// RNNDecoder_44581760532617
// MI455X (gfx1250) — matrix operand order repaired and checked by reading, not compiled, not run on hardware
//
#include <hip/hip_runtime.h>
#include <hip/hip_bf16.h>

typedef __attribute__((ext_vector_type(16))) _Float16 v16h;
typedef __attribute__((ext_vector_type(8)))  _Float16 v8h;
typedef __attribute__((ext_vector_type(8)))  float    v8f;

#define BB   128
#define SS   256
#define RH   512
#define HD   512
#define INP  128
#define G4   (4 * RH)   // 2048

// ---------------------------------------------------------------------------
// Prep: convert weights f32->f16 into workspace, combine LSTM biases.
// ---------------------------------------------------------------------------
#define N_WA1  (RH * RH)          // 262144
#define N_WC   (HD * RH)          // 262144
#define N_WIH  (2 * G4 * RH)      // 2097152
#define N_WL2  (INP * RH)         // 65536
#define N_BIAS (2 * G4)           // 4096
#define N_PREP (N_WA1 + N_WC + N_WIH + N_WL2 + N_BIAS)

__global__ void prep_kernel(const float* Wa1, const float* Wc, const float* W_ih,
                            const float* W_l2, const float* b_ih, const float* b_hh,
                            _Float16* wa1h, _Float16* wch, _Float16* wih0h,
                            _Float16* wih1h, _Float16* wl2h,
                            float* bias0, float* bias1) {
    int i = blockIdx.x * blockDim.x + threadIdx.x;
    if (i < N_WA1) {
        wa1h[i] = (_Float16)Wa1[i];
    } else if (i < N_WA1 + N_WC) {
        int j = i - N_WA1;
        wch[j] = (_Float16)Wc[j];
    } else if (i < N_WA1 + N_WC + N_WIH) {
        int j = i - (N_WA1 + N_WC);
        if (j < G4 * RH) wih0h[j] = (_Float16)W_ih[j];
        else             wih1h[j - G4 * RH] = (_Float16)W_ih[j];
    } else if (i < N_WA1 + N_WC + N_WIH + N_WL2) {
        int j = i - (N_WA1 + N_WC + N_WIH);
        wl2h[j] = (_Float16)W_l2[j];
    } else if (i < N_PREP) {
        int j = i - (N_WA1 + N_WC + N_WIH + N_WL2);
        int l = j >> 11, r = j & (G4 - 1);
        float v = b_ih[l * G4 + r] + b_hh[l * G4 + r];
        if (l == 0) bias0[r] = v; else bias1[r] = v;
    }
}

// ---------------------------------------------------------------------------
// Fused attention-score kernel:
//   score[bs] = sum_n tanh(enc_row . Wa1[n] + ba1[n]) * Wa2[n]
// One block = 32 rows of flattened (b,s) (two 16-row M tiles sharing each B
// fragment -> halves Wa1 L2 traffic). 4 waves split the N (=512) dimension.
// Loop order: k outer, 8 n-tiles inner (16 live accumulators) so the 8
// independent B load-pairs per k-step can be clause'd / pipelined.
// The [B,S,RH] hidden tensor never touches memory.
// ---------------------------------------------------------------------------
#define A_PITCH (RH + 8)   // pad to spread LDS banks (1040 B row stride)
#define MROWS   32
#define NT_PER_WAVE 8      // (RH/16) / 4 waves

__global__ __launch_bounds__(128) void attn_scores_kernel(
        const float* __restrict__ enc, const _Float16* __restrict__ wa1h,
        const float* __restrict__ ba1, const float* __restrict__ wa2,
        float* __restrict__ scores) {
    __shared__ _Float16 As[MROWS][A_PITCH];
    __shared__ float sc[MROWS];

    const int t   = threadIdx.x;
    const int bs0 = blockIdx.x * MROWS;

    if (t < MROWS) sc[t] = 0.0f;

    // Stage 32 encoder rows into LDS as f16 (coalesced f32 reads, one pass).
    const float* ebase = enc + (size_t)bs0 * RH;
    for (int j = t; j < MROWS * RH; j += 128) {
        int r = j >> 9, k = j & (RH - 1);
        As[r][k] = (_Float16)ebase[r * RH + k];
    }
    __syncthreads();

    const int wave = t >> 5;
    const int lane = t & 31;
    const int half = lane >> 4;
    const int l15  = lane & 15;

    const _Float16* arow0 = &As[l15][half ? 8 : 0];        // M tile 0: rows 0..15
    const _Float16* arow1 = &As[16 + l15][half ? 8 : 0];   // M tile 1: rows 16..31

    // B row pointers for this wave's 8 n-tiles.
    const _Float16* brows[NT_PER_WAVE];
#pragma unroll
    for (int tt = 0; tt < NT_PER_WAVE; ++tt) {
        const int n = (((wave + 4 * tt) << 4) + l15);
        brows[tt] = wa1h + (size_t)n * RH + (half ? 8 : 0);
    }

    v8f c0[NT_PER_WAVE];
    v8f c1[NT_PER_WAVE];
#pragma unroll
    for (int tt = 0; tt < NT_PER_WAVE; ++tt) { c0[tt] = (v8f){}; c1[tt] = (v8f){}; }

    for (int k0 = 0; k0 < RH; k0 += 32) {
        // A fragments (ISA 16-bit A 16x32 layout) from LDS -- one pair per k-step.
        v8h a0lo = *(const v8h*)(arow0 + k0);
        v8h a0hi = *(const v8h*)(arow0 + k0 + 16);
        v8h a1lo = *(const v8h*)(arow1 + k0);
        v8h a1hi = *(const v8h*)(arow1 + k0 + 16);
        v16h a0, a1;
#pragma unroll
        for (int i = 0; i < 8; ++i) {
            a0[i] = a0lo[i]; a0[8 + i] = a0hi[i];
            a1[i] = a1lo[i]; a1[8 + i] = a1hi[i];
        }
#pragma unroll
        for (int tt = 0; tt < NT_PER_WAVE; ++tt) {
            // B fragment (same K layout as A: two 8-K runs 16 apart) -- 8 independent
            // load-pairs per k-step, each reused by both M tiles.
            v8h blo = *(const v8h*)(brows[tt] + k0);
            v8h bhi = *(const v8h*)(brows[tt] + k0 + 16);
            v16h b;
#pragma unroll
            for (int i = 0; i < 8; ++i) { b[i] = blo[i]; b[8 + i] = bhi[i]; }
            c0[tt] = __builtin_amdgcn_wmma_f32_16x16x32_f16(false, a0, false, b,
                                                            (short)0, c0[tt], false, false);
            c1[tt] = __builtin_amdgcn_wmma_f32_16x16x32_f16(false, a1, false, b,
                                                            (short)0, c1[tt], false, false);
        }
    }

    // Epilogue: tanh(.+ba1[n]) * Wa2[n], accumulate per-row partials.
    float part0[8] = {0.f, 0.f, 0.f, 0.f, 0.f, 0.f, 0.f, 0.f};
    float part1[8] = {0.f, 0.f, 0.f, 0.f, 0.f, 0.f, 0.f, 0.f};
#pragma unroll
    for (int tt = 0; tt < NT_PER_WAVE; ++tt) {
        const int n = (((wave + 4 * tt) << 4) + l15);
        const float bias = ba1[n];
        const float wv   = wa2[n];
#pragma unroll
        for (int j = 0; j < 8; ++j) {
            part0[j] += tanhf(c0[tt][j] + bias) * wv;
            part1[j] += tanhf(c1[tt][j] + bias) * wv;
        }
    }

    // Butterfly-reduce over the 16 lanes of each half-wave (masks < 16 keep halves apart).
#pragma unroll
    for (int m = 8; m >= 1; m >>= 1)
#pragma unroll
        for (int j = 0; j < 8; ++j) {
            part0[j] += __shfl_xor(part0[j], m, 32);
            part1[j] += __shfl_xor(part1[j], m, 32);
        }

    if (l15 == 0) {
#pragma unroll
        for (int j = 0; j < 8; ++j) {
            atomicAdd(&sc[(half << 3) + j], part0[j]);        // ds_add_f32
            atomicAdd(&sc[16 + (half << 3) + j], part1[j]);
        }
    }
    __syncthreads();
    if (t < MROWS) scores[bs0 + t] = sc[t];
}

// ---------------------------------------------------------------------------
// Softmax over S per batch + weighted encoder sum -> context (f16).
// ---------------------------------------------------------------------------
__global__ __launch_bounds__(256) void softmax_ctx_kernel(
        const float* __restrict__ scores, const float* __restrict__ enc,
        _Float16* __restrict__ ctx) {
    __shared__ float w[SS];
    __shared__ float red[SS];
    const int b = blockIdx.x;
    const int t = threadIdx.x;

    float s = scores[b * SS + t];
    red[t] = s;
    __syncthreads();
    for (int off = 128; off; off >>= 1) {
        if (t < off) red[t] = fmaxf(red[t], red[t + off]);
        __syncthreads();
    }
    const float mx = red[0];
    __syncthreads();
    float e = __expf(s - mx);
    red[t] = e;
    __syncthreads();
    for (int off = 128; off; off >>= 1) {
        if (t < off) red[t] += red[t + off];
        __syncthreads();
    }
    const float inv = 1.0f / red[0];
    __syncthreads();
    w[t] = e * inv;
    __syncthreads();

    const float* ep = enc + (size_t)b * SS * RH;
    for (int d = t; d < RH; d += 256) {
        float acc = 0.0f;
        for (int s2 = 0; s2 < SS; ++s2)
            acc += w[s2] * ep[(size_t)s2 * RH + d];
        ctx[b * RH + d] = (_Float16)acc;
    }
}

// ---------------------------------------------------------------------------
// Generic 1-wave WMMA GEMM: C[M,N] = A[M,K](f16) * W[N,K](f16)^T + bias.
// EPI 0: tanh -> f16 out.  EPI 1: raw f32 out.
// ---------------------------------------------------------------------------
template <int EPI>
__global__ __launch_bounds__(32) void wmma_gemm_kernel(
        const _Float16* __restrict__ A, const _Float16* __restrict__ W,
        const float* __restrict__ bias, void* __restrict__ out,
        int M, int N, int K) {
    const int n0 = blockIdx.x << 4;
    const int m0 = blockIdx.y << 4;
    const int lane = threadIdx.x;
    const int half = lane >> 4;
    const int l15  = lane & 15;

    v8f c = {};
    const _Float16* arow = A + (size_t)(m0 + l15) * K + (half ? 8 : 0);
    const _Float16* brow = W + (size_t)(n0 + l15) * K + (half ? 8 : 0);
    for (int k0 = 0; k0 < K; k0 += 32) {
        v8h alo = *(const v8h*)(arow + k0);
        v8h ahi = *(const v8h*)(arow + k0 + 16);
        v8h blo = *(const v8h*)(brow + k0);
        v8h bhi = *(const v8h*)(brow + k0 + 16);
        v16h a, b;
#pragma unroll
        for (int i = 0; i < 8; ++i) {
            a[i] = alo[i]; a[8 + i] = ahi[i];
            b[i] = blo[i]; b[8 + i] = bhi[i];
        }
        c = __builtin_amdgcn_wmma_f32_16x16x32_f16(false, a, false, b,
                                                   (short)0, c, false, false);
    }
    const int col = n0 + l15;
    const float bv = bias ? bias[col] : 0.0f;
#pragma unroll
    for (int j = 0; j < 8; ++j) {
        const int row = m0 + (half << 3) + j;
        const float v = c[j] + bv;
        if (EPI == 0)
            ((_Float16*)out)[(size_t)row * N + col] = (_Float16)tanhf(v);
        else
            ((float*)out)[(size_t)row * N + col] = v;
    }
}

// ---------------------------------------------------------------------------
// LSTM single-step activation (h0=c0=0): x = sigmoid(o)*tanh(sigmoid(i)*tanh(g))
// gates layout [B, 4*RH] in torch order i,f,g,o. f is dead (c0=0).
// ---------------------------------------------------------------------------
__global__ __launch_bounds__(256) void lstm_act_kernel(
        const float* __restrict__ gates, _Float16* __restrict__ xout) {
    int idx = blockIdx.x * blockDim.x + threadIdx.x;
    if (idx >= BB * RH) return;
    int b = idx >> 9, r = idx & (RH - 1);
    const float* g = gates + ((size_t)b << 11);
    float gi = g[r], gg = g[2 * RH + r], go = g[3 * RH + r];
    float si = 1.0f / (1.0f + __expf(-gi));
    float so = 1.0f / (1.0f + __expf(-go));
    float cc = si * tanhf(gg);
    xout[idx] = (_Float16)(so * tanhf(cc));
}

// ---------------------------------------------------------------------------
// Broadcast y[B,INP] over S -> out[B,S,INP] (float4 stores, 16 MB).
// ---------------------------------------------------------------------------
__global__ __launch_bounds__(256) void bcast_kernel(const float* __restrict__ y,
                                                    float* __restrict__ out) {
    int i4 = blockIdx.x * blockDim.x + threadIdx.x;   // over B*S*INP/4 float4s
    if (i4 >= BB * SS * (INP / 4)) return;
    int b  = i4 >> 13;            // S * INP/4 = 8192 float4 per batch
    int c4 = i4 & (INP / 4 - 1);  // 32 float4 per row
    float4 v = ((const float4*)y)[(b << 5) + c4];
    ((float4*)out)[i4] = v;
}

// ---------------------------------------------------------------------------
// Launcher
// ---------------------------------------------------------------------------
extern "C" void kernel_launch(void* const* d_in, const int* in_sizes, int n_in,
                              void* d_out, int out_size, void* d_ws, size_t ws_size,
                              hipStream_t stream) {
    (void)in_sizes; (void)n_in; (void)out_size; (void)ws_size;
    const float* enc  = (const float*)d_in[1];
    const float* Wa1  = (const float*)d_in[4];
    const float* ba1  = (const float*)d_in[5];
    const float* Wa2  = (const float*)d_in[6];
    const float* Wc   = (const float*)d_in[8];
    const float* bc   = (const float*)d_in[9];
    const float* W_ih = (const float*)d_in[10];
    const float* b_ih = (const float*)d_in[12];
    const float* b_hh = (const float*)d_in[13];
    const float* W_l2 = (const float*)d_in[14];
    const float* b_l2 = (const float*)d_in[15];
    float* out = (float*)d_out;

    char* ws = (char*)d_ws;
    size_t off = 0;
    auto alloc = [&](size_t bytes) { char* p = ws + off; off += (bytes + 255) & ~(size_t)255; return p; };
    _Float16* wa1h  = (_Float16*)alloc(N_WA1 * 2);
    _Float16* wch   = (_Float16*)alloc(N_WC * 2);
    _Float16* wih0h = (_Float16*)alloc((size_t)G4 * RH * 2);
    _Float16* wih1h = (_Float16*)alloc((size_t)G4 * RH * 2);
    _Float16* wl2h  = (_Float16*)alloc(N_WL2 * 2);
    float*    bias0 = (float*)alloc(G4 * 4);
    float*    bias1 = (float*)alloc(G4 * 4);
    float*    scores= (float*)alloc((size_t)BB * SS * 4);
    _Float16* ctxh  = (_Float16*)alloc((size_t)BB * RH * 2);
    _Float16* x1h   = (_Float16*)alloc((size_t)BB * RH * 2);
    float*    gates = (float*)alloc((size_t)BB * G4 * 4);
    _Float16* x2h   = (_Float16*)alloc((size_t)BB * RH * 2);
    _Float16* x3h   = (_Float16*)alloc((size_t)BB * RH * 2);
    float*    y     = (float*)alloc((size_t)BB * INP * 4);

    // 1) weight prep (f32 -> f16, bias combine)
    prep_kernel<<<(N_PREP + 255) / 256, 256, 0, stream>>>(
        Wa1, Wc, W_ih, W_l2, b_ih, b_hh,
        wa1h, wch, wih0h, wih1h, wl2h, bias0, bias1);

    // 2) fused attention scores (17-GFLOP WMMA GEMM; 32 rows/block for B reuse)
    attn_scores_kernel<<<(BB * SS) / MROWS, 128, 0, stream>>>(enc, wa1h, ba1, Wa2, scores);

    // 3) softmax + context
    softmax_ctx_kernel<<<BB, 256, 0, stream>>>(scores, enc, ctxh);

    // 4) x1 = tanh(context @ Wc^T + bc)
    wmma_gemm_kernel<0><<<dim3(HD / 16, BB / 16), 32, 0, stream>>>(
        ctxh, wch, bc, x1h, BB, HD, RH);

    // 5) LSTM layer 0
    wmma_gemm_kernel<1><<<dim3(G4 / 16, BB / 16), 32, 0, stream>>>(
        x1h, wih0h, bias0, gates, BB, G4, HD);
    lstm_act_kernel<<<(BB * RH) / 256, 256, 0, stream>>>(gates, x2h);

    // 6) LSTM layer 1
    wmma_gemm_kernel<1><<<dim3(G4 / 16, BB / 16), 32, 0, stream>>>(
        x2h, wih1h, bias1, gates, BB, G4, RH);
    lstm_act_kernel<<<(BB * RH) / 256, 256, 0, stream>>>(gates, x3h);

    // 7) y = x3 @ W_l2^T + b_l2
    wmma_gemm_kernel<1><<<dim3(INP / 16, BB / 16), 32, 0, stream>>>(
        x3h, wl2h, b_l2, y, BB, INP, RH);

    // 8) broadcast over sequence
    bcast_kernel<<<(BB * SS * (INP / 4)) / 256, 256, 0, stream>>>(y, out);
}
